// ResidualConvUnit_69226282877540
// MI455X (gfx1250) — 8-bit matrix operand order repaired and checked by reading, compiled, not run on hardware
//
#include <hip/hip_runtime.h>
#include <stdint.h>


#define BB 8
#define CC 256
#define HH 96
#define WW 96
#define HP 98
#define WP 98
#define PIX (HH * WW)
#define MTOT (BB * PIX)

#define LDT 80
#define TILE_BYTES (128 * LDT)

typedef int v8i __attribute__((ext_vector_type(8)));

#if defined(__has_builtin)
#if __has_builtin(__builtin_amdgcn_global_load_async_to_lds_b128) && \
    __has_builtin(__builtin_amdgcn_s_wait_asynccnt)
#define HAVE_ASYNC 1
#endif
#endif
#ifndef HAVE_ASYNC
#define HAVE_ASYNC 0
#endif

#if HAVE_ASYNC
#define NSTAGE 3
#else
#define NSTAGE 2
#endif

__device__ __forceinline__ signed char sgn8(float v) {
    return (signed char)((v > 0.0f) - (v < 0.0f));
}

__device__ __forceinline__ void tileOffsets(int kk, int& aOff, int& bOff) {
    int pos = kk >> 2;
    int ky  = (pos * 11) >> 5;
    int kx  = pos - 3 * ky;
    int cs  = (kk & 3) * 64;
    aOff = (ky * WP + kx) * CC + cs;
    bOff = pos * (CC * CC) + cs;
}

#if HAVE_ASYNC
typedef int vsi4 __attribute__((vector_size(4 * sizeof(int))));
typedef __attribute__((address_space(1))) vsi4 g_vsi4;
typedef __attribute__((address_space(3))) vsi4 l_vsi4;

__device__ __forceinline__ void asyncCopy32(const signed char* g, signed char* l) {
    __builtin_amdgcn_global_load_async_to_lds_b128((g_vsi4*)g, (l_vsi4*)l, 0, 0);
    __builtin_amdgcn_global_load_async_to_lds_b128((g_vsi4*)g, (l_vsi4*)l, 16, 0);
}
#endif

#define TSTR 272
__global__ __launch_bounds__(256)
void prep_act(const float* __restrict__ x, const float* __restrict__ mv,
              const float* __restrict__ al, signed char* __restrict__ act) {
    __shared__ __align__(16) signed char T[32 * TSTR];
    const int t  = threadIdx.x;
    const int x0 = blockIdx.x * 32;
    const int y  = blockIdx.y;
    const int b  = blockIdx.z;
    const int xw = t & 31;
    const int c0 = t >> 5;
    #pragma unroll
    for (int i = 0; i < 32; ++i) {
        int c = i * 8 + c0;
        float v = x[((b * CC + c) * HH + y) * WW + x0 + xw] + mv[c];
        float z = (v >= 0.0f) ? v : al[c] * v;
        T[xw * TSTR + c] = sgn8(z);
    }
    __syncthreads();
    const int pix = t >> 3;
    const int off = (t & 7) * 32;
    signed char* dst = act + ((b * HP + y + 1) * WP + (x0 + pix + 1)) * CC + off;
    uint4 d0 = *(const uint4*)&T[pix * TSTR + off];
    uint4 d1 = *(const uint4*)&T[pix * TSTR + off + 16];
    *(uint4*)(dst)      = d0;
    *(uint4*)(dst + 16) = d1;
}

__global__ __launch_bounds__(256)
void pack_w(const float* __restrict__ w, signed char* __restrict__ wp,
            float* __restrict__ scale) {
    const int o = blockIdx.x;
    const int t = threadIdx.x;
    __shared__ float red[256];
    float s = 0.0f;
    for (int e = t; e < CC * 9; e += 256) {
        int c = e / 9, kk = e % 9;
        float v = w[(o * CC + c) * 9 + kk];
        s += fabsf(v);
        wp[(kk * CC + o) * CC + c] = sgn8(v);
    }
    red[t] = s;
    __syncthreads();
    for (int off = 128; off > 0; off >>= 1) {
        if (t < off) red[t] += red[t + off];
        __syncthreads();
    }
    if (t == 0) scale[o] = red[0] * (1.0f / (CC * 9));
}

__device__ __forceinline__ void computeTile(const signed char* stage, int wm, int wn,
                                            int half, int l15, v8i acc[2][4]) {
    const signed char* As_ = stage;
    const signed char* Bs_ = stage + TILE_BYTES;
    v8i afrag[2];
    #pragma unroll
    for (int mi = 0; mi < 2; ++mi) {
        const int row = wm * 32 + mi * 16 + l15;
        #pragma unroll
        for (int p4 = 0; p4 < 4; ++p4) {
            uint2 d = *(const uint2*)&As_[row * LDT + (p4 >> 1) * 32 + half * 16 + (p4 & 1) * 8];
            afrag[mi][2 * p4]     = (int)d.x;
            afrag[mi][2 * p4 + 1] = (int)d.y;
        }
    }
    #pragma unroll
    for (int ni = 0; ni < 4; ++ni) {
        const int col = wn * 64 + ni * 16 + l15;
        uint4 lo = *(const uint4*)&Bs_[col * LDT + half * 16];
        uint4 hi = *(const uint4*)&Bs_[col * LDT + 32 + half * 16];
        v8i bfrag;
        bfrag[0] = (int)lo.x; bfrag[1] = (int)lo.y;
        bfrag[2] = (int)lo.z; bfrag[3] = (int)lo.w;
        bfrag[4] = (int)hi.x; bfrag[5] = (int)hi.y;
        bfrag[6] = (int)hi.z; bfrag[7] = (int)hi.w;
        #pragma unroll
        for (int mi = 0; mi < 2; ++mi) {
            acc[mi][ni] = __builtin_amdgcn_wmma_i32_16x16x64_iu8(
                true, afrag[mi], true, bfrag, acc[mi][ni], false, false);
        }
    }
}

template <bool FINAL>
__global__ __launch_bounds__(256, 1)
void bconv(const signed char* __restrict__ actIn,
           const signed char* __restrict__ wp,
           const float* __restrict__ scale, const float* __restrict__ bias,
           const float* __restrict__ mv, const float* __restrict__ al,
           signed char* __restrict__ actOut, float* __restrict__ out) {
    __shared__ __align__(16) signed char smem[NSTAGE * 2 * TILE_BYTES];

    const int tid    = threadIdx.x;
    const int lane   = tid & 31;
    const int wv     = tid >> 5;
    const int wm     = wv & 3;
    const int wn     = wv >> 2;
    const int blockM = blockIdx.x;
    const int blockN = blockIdx.y;
    const int half   = lane >> 4;
    const int l15    = lane & 15;

    const int ldRow = tid >> 1;
    const int ldOff = (tid & 1) * 32;

    const int mG   = blockM * 128 + ldRow;
    const int bImg = mG / PIX;
    const int rem  = mG - bImg * PIX;
    const int py   = rem / WW;
    const int px   = rem - py * WW;
    const int aThreadBase = bImg * (HP * WP * CC) + (py * WP + px) * CC + ldOff;
    const int bThreadBase = (blockN * 128 + ldRow) * CC + ldOff;

    v8i acc[2][4];
    {
        v8i z = {};
        #pragma unroll
        for (int i = 0; i < 2; ++i)
            #pragma unroll
            for (int j = 0; j < 4; ++j) acc[i][j] = z;
    }

#if HAVE_ASYNC
    auto issueTile = [&](int kk, signed char* stage) {
        int aOff, bOff;
        tileOffsets(kk, aOff, bOff);
        asyncCopy32(actIn + aThreadBase + aOff, stage + ldRow * LDT + ldOff);
        asyncCopy32(wp + bThreadBase + bOff, stage + TILE_BYTES + ldRow * LDT + ldOff);
    };
    signed char* st0 = smem;
    signed char* st1 = smem + 2 * TILE_BYTES;
    signed char* st2 = smem + 4 * TILE_BYTES;
    issueTile(0, st0);
    issueTile(1, st1);
    __builtin_amdgcn_s_wait_asynccnt(4);
    __syncthreads();
    for (int kk = 0; kk < 34; ++kk) {
        issueTile(kk + 2, st2);
        computeTile(st0, wm, wn, half, l15, acc);
        __builtin_amdgcn_s_wait_asynccnt(4);
        __syncthreads();
        signed char* t = st0; st0 = st1; st1 = st2; st2 = t;
    }
    computeTile(st0, wm, wn, half, l15, acc);
    __builtin_amdgcn_s_wait_asynccnt(0);
    __syncthreads();
    { signed char* t = st0; st0 = st1; st1 = st2; st2 = t; }
    computeTile(st0, wm, wn, half, l15, acc);
#else
    auto loadRegs = [&](int kk, uint4 r[4]) {
        int aOff, bOff;
        tileOffsets(kk, aOff, bOff);
        const uint4* pa = (const uint4*)(actIn + aThreadBase + aOff);
        const uint4* pb = (const uint4*)(wp + bThreadBase + bOff);
        r[0] = pa[0]; r[1] = pa[1]; r[2] = pb[0]; r[3] = pb[1];
    };
    auto storeRegs = [&](const uint4 r[4], signed char* stage) {
        *(uint4*)(stage + ldRow * LDT + ldOff)                   = r[0];
        *(uint4*)(stage + ldRow * LDT + ldOff + 16)              = r[1];
        *(uint4*)(stage + TILE_BYTES + ldRow * LDT + ldOff)      = r[2];
        *(uint4*)(stage + TILE_BYTES + ldRow * LDT + ldOff + 16) = r[3];
    };
    signed char* buf0 = smem;
    signed char* buf1 = smem + 2 * TILE_BYTES;
    uint4 regs[4];
    loadRegs(0, regs);
    storeRegs(regs, buf0);
    loadRegs(1, regs);
    __syncthreads();
    for (int kk = 0; kk < 36; ++kk) {
        signed char* cur = (kk & 1) ? buf1 : buf0;
        signed char* nxt = (kk & 1) ? buf0 : buf1;
        computeTile(cur, wm, wn, half, l15, acc);
        if (kk < 35) storeRegs(regs, nxt);
        if (kk < 34) loadRegs(kk + 2, regs);
        __syncthreads();
    }
#endif

    int mB[2], mY[2], mX[2];
    #pragma unroll
    for (int mi = 0; mi < 2; ++mi) {
        int mBase = blockM * 128 + wm * 32 + mi * 16 + half * 8;
        int b  = mBase / PIX;
        int rr = mBase - b * PIX;
        int y  = rr / WW;
        mB[mi] = b; mY[mi] = y; mX[mi] = rr - y * WW;
    }

    if (FINAL) {
        #pragma unroll
        for (int ni = 0; ni < 4; ++ni) {
            const int nG = blockN * 128 + wn * 64 + ni * 16 + l15;
            const float sc = scale[nG];
            const float bi = bias[nG];
            #pragma unroll
            for (int mi = 0; mi < 2; ++mi) {
                float4 f0, f1;
                f0.x = sc * (float)acc[mi][ni][0] + bi;
                f0.y = sc * (float)acc[mi][ni][1] + bi;
                f0.z = sc * (float)acc[mi][ni][2] + bi;
                f0.w = sc * (float)acc[mi][ni][3] + bi;
                f1.x = sc * (float)acc[mi][ni][4] + bi;
                f1.y = sc * (float)acc[mi][ni][5] + bi;
                f1.z = sc * (float)acc[mi][ni][6] + bi;
                f1.w = sc * (float)acc[mi][ni][7] + bi;
                float* dst = out + ((mB[mi] * CC + nG) * HH + mY[mi]) * WW + mX[mi];
                *(float4*)(dst)     = f0;
                *(float4*)(dst + 4) = f1;
            }
        }
    } else {
        __syncthreads();
        signed char* T = smem;
        #pragma unroll
        for (int ni = 0; ni < 4; ++ni) {
            const int chanLocal = wn * 64 + ni * 16 + l15;
            const int nG = blockN * 128 + chanLocal;
            const float sc = scale[nG];
            const float bi = bias[nG];
            const float m2v = mv[nG];
            const float a2v = al[nG];
            #pragma unroll
            for (int mi = 0; mi < 2; ++mi) {
                const int pixLocal = wm * 32 + mi * 16 + half * 8;
                #pragma unroll
                for (int r = 0; r < 8; ++r) {
                    float v = sc * (float)acc[mi][ni][r] + bi;
                    float t = v + m2v;
                    float z = (t >= 0.0f) ? t : a2v * t;
                    T[(pixLocal + r) * 128 + chanLocal] = sgn8(z);
                }
            }
        }
        __syncthreads();
        const int pix = tid >> 1;
        const int off = (tid & 1) * 64;
        const int pixG = blockM * 128 + pix;
        const int b  = pixG / PIX;
        const int rr = pixG - b * PIX;
        const int y  = rr / WW;
        const int x  = rr - y * WW;
        signed char* dst =
            actOut + ((b * HP + y + 1) * WP + x + 1) * CC + blockN * 128 + off;
        const uint4* src = (const uint4*)&T[pix * 128 + off];
        uint4 d0 = src[0], d1 = src[1], d2 = src[2], d3 = src[3];
        *(uint4*)(dst)      = d0;
        *(uint4*)(dst + 16) = d1;
        *(uint4*)(dst + 32) = d2;
        *(uint4*)(dst + 48) = d3;
    }
}

extern "C" void kernel_launch(void* const* d_in, const int* in_sizes, int n_in,
                              void* d_out, int out_size, void* d_ws, size_t ws_size,
                              hipStream_t stream) {
    const float* x  = (const float*)d_in[0];
    const float* w1 = (const float*)d_in[1];
    const float* b1 = (const float*)d_in[2];
    const float* w2 = (const float*)d_in[3];
    const float* b2 = (const float*)d_in[4];
    const float* a1 = (const float*)d_in[5];
    const float* a2 = (const float*)d_in[6];
    const float* m1 = (const float*)d_in[7];
    const float* m2 = (const float*)d_in[8];

    const size_t actBytes = (size_t)BB * HP * WP * CC;
    const size_t wpBytes  = (size_t)9 * CC * CC;

    char* ws = (char*)d_ws;
    signed char* act0 = (signed char*)(ws);
    signed char* act1 = (signed char*)(ws + actBytes);
    signed char* w1p  = (signed char*)(ws + 2 * actBytes);
    signed char* w2p  = (signed char*)(ws + 2 * actBytes + wpBytes);
    float*       sc1  = (float*)(ws + 2 * actBytes + 2 * wpBytes);
    float*       sc2  = sc1 + CC;

    (void)hipMemsetAsync(act0, 0, actBytes, stream);
    (void)hipMemsetAsync(act1, 0, actBytes, stream);

    prep_act<<<dim3(WW / 32, HH, BB), 256, 0, stream>>>(x, m1, a1, act0);
    pack_w<<<CC, 256, 0, stream>>>(w1, w1p, sc1);
    pack_w<<<CC, 256, 0, stream>>>(w2, w2p, sc2);

    dim3 grid(MTOT / 128, CC / 128);
    bconv<false><<<grid, 256, 0, stream>>>(act0, w1p, sc1, b1, m2, a2, act1, nullptr);
    bconv<true><<<grid, 256, 0, stream>>>(act1, w2p, sc2, b2, nullptr, nullptr,
                                          nullptr, (float*)d_out);
}
